// ResNet1D_1236950581283
// MI455X (gfx1250) — matrix operand order repaired and checked by reading, not compiled, not run on hardware
//
#include <hip/hip_runtime.h>
#include <hip/hip_bf16.h>

typedef __attribute__((ext_vector_type(16))) __bf16 v16bf;
typedef __attribute__((ext_vector_type(8)))  __bf16 v8bf;
typedef __attribute__((ext_vector_type(16))) unsigned short v16u;
typedef __attribute__((ext_vector_type(8)))  float v8f;

// ---------- helpers ----------
__device__ __forceinline__ unsigned short f2bf(float f) {
    __bf16 h = (__bf16)f;                                  // hw v_cvt to bf16 (RNE)
    return __builtin_bit_cast(unsigned short, h);
}
__device__ __forceinline__ v16bf cvt16(v8f lo, v8f hi) {   // packed f32->bf16
    v8bf a = __builtin_convertvector(lo, v8bf);
    v8bf b = __builtin_convertvector(hi, v8bf);
    return __builtin_shufflevector(a, b, 0, 1, 2, 3, 4, 5, 6, 7,
                                   8, 9, 10, 11, 12, 13, 14, 15);
}
__device__ __forceinline__ float silu(float v) { return v / (1.f + __expf(-v)); }

#define WMMA_BF16(A, B, C) \
    __builtin_amdgcn_wmma_f32_16x16x32_bf16(false, (A), false, (B), (short)0, (C), false, false)

// =====================================================================
// Implicit-GEMM Conv1d + (bias) + BN + (residual) + (ReLU), bf16 WMMA.
// Block: 256 threads = 8 waves, computes [128 co x 64 l] for one batch.
//   grid.x = Lout/64, grid.y = ceil(Cout/16/8), grid.z = B
// Per K-step (32): all threads cooperatively stage the 32K x 64N input
// tile (bf16, fragment-swizzled) in LDS; each wave then does 1 A-fragment
// load + 4 LDS B-fragment loads + 4 WMMAs.
// =====================================================================
template<int KW>
__global__ void conv1d_bn_wmma(
    const float* __restrict__ x, const float* __restrict__ w,
    const float* __restrict__ cbias,
    const float* __restrict__ bng, const float* __restrict__ bnb,
    const float* __restrict__ bnm, const float* __restrict__ bnv,
    const float* __restrict__ res, float* __restrict__ out,
    int Cin, int Lin, int Cout, int Lout, int stride, int pad, int relu)
{
    __shared__ __align__(32) unsigned short Bs[32 * 64];   // 4 KB: [tile][lane][slot]

    const int tid  = threadIdx.x;
    const int lane = tid & 31;
    const int wave = tid >> 5;
    const int b    = blockIdx.z;
    const int l0   = blockIdx.x << 6;                      // 64 outputs per block
    const int co0  = (blockIdx.y * 8 + wave) << 4;
    const bool active = (co0 < Cout);                      // wave-uniform
    const int Ktot = Cin * KW;

    const int m  = lane & 15;
    const int hi = lane >> 4;

    const float* __restrict__ xb   = x + (size_t)b * Cin * Lin;
    const float* __restrict__ wrow = w + (size_t)(active ? (co0 + m) : 0) * Ktot;

    // cooperative-fill mapping: thread fills k = tid>>3, n = (tid&7)*8 + i
    const int fk  = tid >> 3;
    const int fn0 = (tid & 7) << 3;
    const int fslot  = (fk & 7) | ((fk >> 4) << 3);
    const int flhalf = ((fk >> 3) & 1) << 4;

    v8f acc[4] = {};
    for (int kk = 0; kk < Ktot; kk += 32) {
        // ---- stage B tile in LDS ----
        {
            int k  = kk + fk;
            bool kv = (k < Ktot);
            int ci  = k / KW;                              // compile-time KW
            int kwi = k - ci * KW;
            const float* xc = xb + (size_t)ci * Lin;
#pragma unroll
            for (int i = 0; i < 8; ++i) {
                int n = fn0 + i;
                float v = 0.f;
                if (kv) {
                    int pos = (l0 + n) * stride + kwi - pad;
                    if (pos >= 0 && pos < Lin) v = xc[pos];
                }
                int lanep = (n & 15) + flhalf;
                Bs[(((n >> 4) << 5) + lanep) * 16 + fslot] = f2bf(v);
            }
        }
        __syncthreads();
        if (active) {
            // A fragment: weights row co0+m, K offsets per ISA 16-bit A layout
            v16bf a;
            if (kk + 32 <= Ktot) {                         // wave-uniform fast path
                v8f f0 = *(const v8f*)(wrow + kk + hi * 8);
                v8f f1 = *(const v8f*)(wrow + kk + 16 + hi * 8);
                a = cvt16(f0, f1);
            } else {
                v16u au;
#pragma unroll
                for (int e = 0; e < 16; ++e) {
                    int kofs = (e < 8) ? (hi * 8 + e) : (16 + hi * 8 + (e - 8));
                    int k = kk + kofs;
                    au[e] = (k < Ktot) ? f2bf(wrow[k]) : (unsigned short)0;
                }
                a = __builtin_bit_cast(v16bf, au);
            }
            const v16u* Bv = reinterpret_cast<const v16u*>(Bs);
#pragma unroll
            for (int t = 0; t < 4; ++t) {
                v16bf bf = __builtin_bit_cast(v16bf, Bv[(t << 5) + lane]);
                acc[t] = WMMA_BF16(a, bf, acc[t]);
            }
        }
        __syncthreads();
    }
    if (!active) return;

    // ---- epilogue: co = co0 + r + 8*hi, l = l0 + 16*t + (lane&15) ----
#pragma unroll
    for (int r = 0; r < 8; ++r) {
        int co = co0 + r + hi * 8;
        float scale = bng[co] * rsqrtf(bnv[co] + 1e-5f);
        float shift = bnb[co] - bnm[co] * scale;
        float bias  = cbias ? cbias[co] : 0.f;
        size_t base = ((size_t)b * Cout + co) * Lout;
#pragma unroll
        for (int t = 0; t < 4; ++t) {
            int l = l0 + (t << 4) + m;
            float v = (acc[t][r] + bias) * scale + shift;
            size_t oidx = base + l;
            if (res) v += res[oidx];
            if (relu) v = fmaxf(v, 0.f);
            out[oidx] = v;
        }
    }
}

// =====================================================================
// GEMM: out[r,o] = sum_d X[r,d] * W[o,d]  (X:[R,D], W:[O,D], row-major)
// Wave computes 16 rows x 64 outputs (A fragment reused for 4 WMMAs).
//   grid.x = O/64, grid.y = R/128; block = 256 threads.
// tr_store: out as [B, O, Lseg] with Lseg = 1<<lsh.
// =====================================================================
__global__ void gemm_bt_wmma(const float* __restrict__ X, const float* __restrict__ W,
                             float* __restrict__ out, int D, int O,
                             int tr_store, int lsh)
{
    const int lane = threadIdx.x & 31;
    const int wave = threadIdx.x >> 5;
    const int o0 = blockIdx.x << 6;
    const int r0 = (blockIdx.y * 8 + wave) << 4;
    const int m = lane & 15, hi = lane >> 4;

    const float* __restrict__ xrow = X + (size_t)(r0 + m) * D;

    v8f acc[4] = {};
    for (int kk = 0; kk < D; kk += 32) {
        v8f f0 = *(const v8f*)(xrow + kk + hi * 8);
        v8f f1 = *(const v8f*)(xrow + kk + 16 + hi * 8);
        v16bf a = cvt16(f0, f1);
#pragma unroll
        for (int t = 0; t < 4; ++t) {
            const float* wc = W + (size_t)(o0 + (t << 4) + m) * D + kk + hi * 8;
            v16bf bf = cvt16(*(const v8f*)wc, *(const v8f*)(wc + 16));
            acc[t] = WMMA_BF16(a, bf, acc[t]);
        }
    }
#pragma unroll
    for (int rr = 0; rr < 8; ++rr) {
        int r = r0 + rr + hi * 8;
#pragma unroll
        for (int t = 0; t < 4; ++t) {
            int o = o0 + (t << 4) + m;
            float v = acc[t][rr];
            if (tr_store) {
                size_t oidx = (((size_t)(r >> lsh) * O + o) << lsh) + (r & ((1 << lsh) - 1));
                out[oidx] = v;
            } else {
                out[(size_t)r * O + o] = v;
            }
        }
    }
}

// ---------- LayerNorm over channel dim + transpose [B,C,L] -> [B*L, C] ----------
__global__ void ln_transpose_k(const float* __restrict__ act, const float* __restrict__ g,
                               const float* __restrict__ be, float* __restrict__ t,
                               int Bn, int C, int L)
{
    int r = blockIdx.x * blockDim.x + threadIdx.x;
    if (r >= Bn * L) return;
    int b = r / L, l = r - b * L;
    const float* a = act + (size_t)b * C * L + l;
    float mu = 0.f;
    for (int c = 0; c < C; ++c) mu += a[(size_t)c * L];
    mu /= C;
    float var = 0.f;
    for (int c = 0; c < C; ++c) { float d = a[(size_t)c * L] - mu; var += d * d; }
    float inv = rsqrtf(var / C + 1e-5f);
    float* tr = t + (size_t)r * C;
    for (int c = 0; c < C; ++c) tr[c] = (a[(size_t)c * L] - mu) * inv * g[c] + be[c];
}

// ---------- causal depthwise conv (dc=2) + SiLU; u-half of uz [R, 2D] ----------
__global__ void dwconv_silu_k(const float* __restrict__ uz, const float* __restrict__ cw,
                              const float* __restrict__ cb, float* __restrict__ u,
                              int Bn, int L, int D)
{
    size_t idx = (size_t)blockIdx.x * blockDim.x + threadIdx.x;
    if (idx >= (size_t)Bn * L * D) return;
    int d = idx % D; size_t r = idx / D; int l = (int)(r % L);
    float cur  = uz[r * 2 * D + d];
    float prev = (l > 0) ? uz[(r - 1) * 2 * D + d] : 0.f;
    float v = prev * cw[d * 2 + 0] + cur * cw[d * 2 + 1] + cb[d];
    u[r * D + d] = silu(v);
}

// ---------- x-projection: xp[r,j] = dot(u[r,:], xw[j,:]) ----------
__global__ void xproj_k(const float* __restrict__ u, const float* __restrict__ xw,
                        float* __restrict__ xp, int R, int D, int J)
{
    int idx = blockIdx.x * blockDim.x + threadIdx.x;
    if (idx >= R * J) return;
    int r = idx / J, j = idx - r * J;
    const float* ur = u + (size_t)r * D;
    const float* wr = xw + (size_t)j * D;
    float s = 0.f;
    for (int d = 0; d < D; ++d) s += ur[d] * wr[d];
    xp[idx] = s;
}

// ---------- dt = softplus(xp[:, :dtr] @ dt_w.T + dt_b) ----------
__global__ void dtproj_k(const float* __restrict__ xp, const float* __restrict__ dtw,
                         const float* __restrict__ dtb, float* __restrict__ dt,
                         int R, int D, int dtr, int J)
{
    size_t idx = (size_t)blockIdx.x * blockDim.x + threadIdx.x;
    if (idx >= (size_t)R * D) return;
    int d = idx % D; size_t r = idx / D;
    const float* xr = xp + r * J;
    const float* wr = dtw + (size_t)d * dtr;
    float s = dtb[d];
    for (int j = 0; j < dtr; ++j) s += xr[j] * wr[j];
    dt[idx] = (s > 20.f) ? s : log1pf(__expf(s));
}

// ---------- selective scan: one thread per (b, d), n=4 states ----------
__global__ void mamba_scan_k(const float* __restrict__ dt, const float* __restrict__ xp,
                             const float* __restrict__ u, const float* __restrict__ uz,
                             const float* __restrict__ alog, const float* __restrict__ dp,
                             float* __restrict__ y, int Bn, int L, int D, int J, int dtr)
{
    int idx = blockIdx.x * blockDim.x + threadIdx.x;
    if (idx >= Bn * D) return;
    int b = idx / D, d = idx - b * D;
    float A0 = -__expf(alog[d * 4 + 0]), A1 = -__expf(alog[d * 4 + 1]);
    float A2 = -__expf(alog[d * 4 + 2]), A3 = -__expf(alog[d * 4 + 3]);
    float h0 = 0.f, h1 = 0.f, h2 = 0.f, h3 = 0.f;
    float dpd = dp[d];
    for (int l = 0; l < L; ++l) {
        size_t r = (size_t)b * L + l;
        float dtv = dt[r * D + d];
        float uv  = u[r * D + d];
        const float* xr = xp + r * J + dtr;
        float du = dtv * uv;
        h0 = __expf(dtv * A0) * h0 + du * xr[0];
        h1 = __expf(dtv * A1) * h1 + du * xr[1];
        h2 = __expf(dtv * A2) * h2 + du * xr[2];
        h3 = __expf(dtv * A3) * h3 + du * xr[3];
        float yv = h0 * xr[4] + h1 * xr[5] + h2 * xr[6] + h3 * xr[7] + uv * dpd;
        float zv = uz[r * 2 * D + D + d];
        y[r * D + d] = yv * silu(zv);
    }
}

// ---------- global average pool over L ----------
__global__ void avgpool_k(const float* __restrict__ act, float* __restrict__ pool,
                          int Bn, int C, int L)
{
    int i = blockIdx.x * blockDim.x + threadIdx.x;
    if (i >= Bn * C) return;
    const float* a = act + (size_t)i * L;
    float s = 0.f;
    for (int l = 0; l < L; ++l) s += a[l];
    pool[i] = s / (float)L;
}

// ---------- final linear 1024 -> 4096 ----------
__global__ void linear_k(const float* __restrict__ pool, const float* __restrict__ wt,
                         const float* __restrict__ bs, float* __restrict__ out,
                         int Bn, int Din, int Dout)
{
    int i = blockIdx.x * blockDim.x + threadIdx.x;
    if (i >= Bn * Dout) return;
    int b = i / Dout, o = i - b * Dout;
    const float* p  = pool + (size_t)b * Din;
    const float* wr = wt + (size_t)o * Din;
    float s = bs[o];
    for (int d = 0; d < Din; ++d) s += p[d] * wr[d];
    out[i] = s;
}

// =====================================================================
extern "C" void kernel_launch(void* const* d_in, const int* in_sizes, int n_in,
                              void* d_out, int out_size, void* d_ws, size_t ws_size,
                              hipStream_t stream)
{
    (void)in_sizes; (void)n_in; (void)out_size; (void)ws_size;
    const int Bn = 8;
    int idx = 0;
    auto F = [&](int i) { return (const float*)d_in[i]; };

    // input order: depth-first dict insertion order of setup_inputs()
    const float* x = F(idx++);
    const float* stem_w = F(idx++);
    const float* stem_b = F(idx++);
    const float* sg = F(idx++); const float* sb = F(idx++);
    const float* sm = F(idx++); const float* sv = F(idx++);

    // workspace carve (floats)
    float* ws = (float*)d_ws;
    float* bufs[3] = { ws, ws + (1 << 20), ws + 2 * (1 << 20) };
    float* t    = ws + 3 * (size_t)(1 << 20);
    float* uz   = ws + 4 * (size_t)(1 << 20);   // 2M floats
    float* ub   = ws + 6 * (size_t)(1 << 20);
    float* dt   = ws + 7 * (size_t)(1 << 20);
    float* yb   = ws + 8 * (size_t)(1 << 20);
    float* xp   = ws + 9 * (size_t)(1 << 20);
    float* pool = xp + (1 << 18);

    auto launch_conv = [&](const float* in, const float* w, const float* cb,
                           const float* g, const float* b, const float* m, const float* v,
                           const float* res, float* out,
                           int Cin, int Lin, int Cout, int Lout,
                           int Kw, int stride, int pad, int relu) {
        dim3 grid(Lout / 64, ((Cout >> 4) + 7) / 8, Bn);
        if (Kw == 3)
            conv1d_bn_wmma<3><<<grid, 256, 0, stream>>>(in, w, cb, g, b, m, v, res, out,
                                                        Cin, Lin, Cout, Lout, stride, pad, relu);
        else if (Kw == 1)
            conv1d_bn_wmma<1><<<grid, 256, 0, stream>>>(in, w, cb, g, b, m, v, res, out,
                                                        Cin, Lin, Cout, Lout, stride, pad, relu);
        else
            conv1d_bn_wmma<7><<<grid, 256, 0, stream>>>(in, w, cb, g, b, m, v, res, out,
                                                        Cin, Lin, Cout, Lout, stride, pad, relu);
    };
    auto launch_gemm = [&](const float* X, const float* W, float* out,
                           int R, int D, int O, int tr, int lsh) {
        dim3 grid(O / 64, R / 128);
        gemm_bt_wmma<<<grid, 256, 0, stream>>>(X, W, out, D, O, tr, lsh);
    };

    // ---- stem: conv(1->16, k7, s2, p3) + bias + BN + ReLU ----
    float* cur = bufs[0]; int curBuf = 0;
    launch_conv(x, stem_w, stem_b, sg, sb, sm, sv, nullptr, cur,
                1, 16384, 16, 8192, 7, 2, 3, 1);
    int curC = 16, curL = 8192;

    const int CH[7] = {16, 32, 64, 128, 256, 512, 1024};
    const int ST[7] = {1, 2, 2, 2, 2, 2, 2};

    for (int si = 0; si < 7; ++si) {
        int cout = CH[si];
        for (int blk = 0; blk < 2; ++blk) {
            int s = (blk == 0) ? ST[si] : 1;
            int cin = curC;
            bool has_wd = (s != 1) || (cin != cout);
            const float* w1 = F(idx++);
            const float* g1 = F(idx++); const float* b1 = F(idx++);
            const float* m1 = F(idx++); const float* v1 = F(idx++);
            const float* w2 = F(idx++);
            const float* g2 = F(idx++); const float* b2 = F(idx++);
            const float* m2 = F(idx++); const float* v2 = F(idx++);
            const float *wd = nullptr, *gd = nullptr, *bd = nullptr, *md = nullptr, *vd = nullptr;
            if (has_wd) { wd = F(idx++); gd = F(idx++); bd = F(idx++); md = F(idx++); vd = F(idx++); }
            int Lout = curL / s;

            int iT1 = -1, iSc = -1;
            for (int k = 0; k < 3; ++k)
                if (k != curBuf) { if (iT1 < 0) iT1 = k; else iSc = k; }
            float* bT1 = bufs[iT1];
            float* bSc = bufs[iSc];

            const float* scp; float* outp; int outBuf;
            if (has_wd) {
                launch_conv(cur, wd, nullptr, gd, bd, md, vd, nullptr, bSc,
                            cin, curL, cout, Lout, 1, s, 0, 0);
                scp = bSc; outp = bSc; outBuf = iSc;
            } else { scp = cur; outp = cur; outBuf = curBuf; }

            launch_conv(cur, w1, nullptr, g1, b1, m1, v1, nullptr, bT1,
                        cin, curL, cout, Lout, 3, s, 1, 1);
            // conv2 + BN + residual + ReLU (out aliases res: per-element read-then-write)
            launch_conv(bT1, w2, nullptr, g2, b2, m2, v2, scp, outp,
                        cout, Lout, cout, Lout, 3, 1, 1, 1);
            cur = bufs[outBuf]; curBuf = outBuf; curC = cout; curL = Lout;
        }

        if (si == 3 || si == 5) {
            int D = curC, L = curL, R = Bn * L;
            int dtr = (D + 15) / 16, J = dtr + 8;
            int lsh = (L == 1024) ? 10 : 8;
            const float* lng = F(idx++); const float* lnb = F(idx++);
            const float* in_w = F(idx++); const float* cw  = F(idx++);
            const float* cb2  = F(idx++); const float* xw  = F(idx++);
            const float* dtw  = F(idx++); const float* dtb = F(idx++);
            const float* alog = F(idx++); const float* dp  = F(idx++);
            const float* ow   = F(idx++);

            ln_transpose_k<<<(R + 255) / 256, 256, 0, stream>>>(cur, lng, lnb, t, Bn, D, L);
            launch_gemm(t, in_w, uz, R, D, 2 * D, 0, 0);
            dwconv_silu_k<<<((size_t)R * D + 255) / 256, 256, 0, stream>>>(uz, cw, cb2, ub, Bn, L, D);
            xproj_k<<<(R * J + 255) / 256, 256, 0, stream>>>(ub, xw, xp, R, D, J);
            dtproj_k<<<((size_t)R * D + 255) / 256, 256, 0, stream>>>(xp, dtw, dtb, dt, R, D, dtr, J);
            mamba_scan_k<<<(Bn * D + 255) / 256, 256, 0, stream>>>(dt, xp, ub, uz, alog, dp, yb,
                                                                   Bn, L, D, J, dtr);
            launch_gemm(yb, ow, cur, R, D, D, 1, lsh);
        }
    }

    // ---- average pool over L=128, then linear 1024 -> 4096 ----
    avgpool_k<<<(Bn * 1024 + 255) / 256, 256, 0, stream>>>(cur, pool, Bn, 1024, 128);
    const float* lw = F(idx++);
    const float* lb = F(idx++);
    linear_k<<<(Bn * 4096 + 255) / 256, 256, 0, stream>>>(pool, lw, lb, (float*)d_out,
                                                          Bn, 1024, 4096);
}
